// SAGE_43800076484795
// MI455X (gfx1250) — matrix operand order repaired and checked by reading, not compiled, not run on hardware
//
#include <hip/hip_runtime.h>

#define N_NODES 50000
#define N_EDGES 800000
#define D_FEAT  128
#define D_CLS   47
#define K_TILES 8          // K=256 / 32 (bf16 WMMA K)
#define A_STRIDE 274       // bf16 elems per staged LDS row; 137 dwords -> bank-conflict-free frag gather
#define SCAN_T  1024
#define CHUNK   ((N_NODES + SCAN_T - 1) / SCAN_T)   // 49

typedef __attribute__((ext_vector_type(16))) __bf16 v16bf;
typedef __attribute__((ext_vector_type(8)))  float  v8f;

union Frag16 { v16bf v; unsigned int u[8]; uint4 q[2]; };

__device__ __forceinline__ unsigned short f2bf(float f) {
  unsigned int u = __builtin_bit_cast(unsigned int, f);
  u += 0x7FFFu + ((u >> 16) & 1u);          // round-to-nearest-even
  return (unsigned short)(u >> 16);
}

// ---------------- CSR construction (once per launch) ----------------
__global__ void count_edges_kernel(const long long* __restrict__ dst, int* __restrict__ deg) {
  int e = blockIdx.x * blockDim.x + threadIdx.x;
  if (e < N_EDGES) atomicAdd(&deg[(int)dst[e]], 1);
}

// single-block exclusive scan: 1024 threads x 49-element chunks + LDS Hillis-Steele
__global__ __launch_bounds__(SCAN_T) void scan_offsets_kernel(const int* __restrict__ deg,
                                                              int* __restrict__ offs) {
  __shared__ int sums[SCAN_T];
  int t = threadIdx.x;
  int base = t * CHUNK;
  int s = 0;
  for (int i = 0; i < CHUNK; ++i) {
    int idx = base + i;
    if (idx < N_NODES) s += deg[idx];
  }
  sums[t] = s;
  __syncthreads();
  for (int d = 1; d < SCAN_T; d <<= 1) {
    int v = (t >= d) ? sums[t - d] : 0;
    __syncthreads();
    sums[t] += v;
    __syncthreads();
  }
  int prefix = (t == 0) ? 0 : sums[t - 1];
  for (int i = 0; i < CHUNK; ++i) {
    int idx = base + i;
    if (idx < N_NODES) { offs[idx] = prefix; prefix += deg[idx]; }
  }
  if (t == 0) offs[N_NODES] = N_EDGES;
}

__global__ void copy_int_kernel(const int* __restrict__ in, int* __restrict__ out, int n) {
  int i = blockIdx.x * blockDim.x + threadIdx.x;
  if (i < n) out[i] = in[i];
}

__global__ void fill_csr_kernel(const long long* __restrict__ src,
                                const long long* __restrict__ dst,
                                int* __restrict__ cursor, int* __restrict__ csr) {
  int e = blockIdx.x * blockDim.x + threadIdx.x;
  if (e < N_EDGES) {
    int d = (int)dst[e];
    int pos = atomicAdd(&cursor[d], 1);
    csr[pos] = (int)src[e];
  }
}

// ---------------- pull-mode mean aggregation: one wave per node ----------------
__global__ void aggregate_kernel(const float* __restrict__ feat,
                                 const int* __restrict__ csr,
                                 const int* __restrict__ offs,
                                 float* __restrict__ mean) {
  int node = __builtin_amdgcn_readfirstlane(blockIdx.x * (blockDim.x >> 5) + (threadIdx.x >> 5));
  int lane = threadIdx.x & 31;
  if (node >= N_NODES) return;
  int beg = offs[node], end = offs[node + 1];       // wave-uniform -> scalar loads
  float4 acc = make_float4(0.f, 0.f, 0.f, 0.f);
  int i = beg;
  for (; i + 1 < end; i += 2) {
    int s0 = csr[i];
    int s1 = csr[i + 1];
    float4 v0 = ((const float4*)(feat + (size_t)s0 * D_FEAT))[lane];
    float4 v1 = ((const float4*)(feat + (size_t)s1 * D_FEAT))[lane];
    acc.x += v0.x + v1.x; acc.y += v0.y + v1.y;
    acc.z += v0.z + v1.z; acc.w += v0.w + v1.w;
  }
  if (i < end) {
    int s0 = csr[i];
    float4 v0 = ((const float4*)(feat + (size_t)s0 * D_FEAT))[lane];
    acc.x += v0.x; acc.y += v0.y; acc.z += v0.z; acc.w += v0.w;
  }
  float inv = 1.0f / fmaxf((float)(end - beg), 1.0f);
  acc.x *= inv; acc.y *= inv; acc.z *= inv; acc.w *= inv;
  ((float4*)(mean + (size_t)node * D_FEAT))[lane] = acc;
}

// ---------------- pre-swizzle weights into per-lane WMMA B-fragments ----------------
// B = [Wl ; Wr] (256 x dout). Fragment (ktile, ntile): lane l holds column
// ntile*16 + (l&15), K range ktile*32 + (l>>4)*16 + i, packed 2 bf16 per VGPR.
__global__ void pack_weights_kernel(const float* __restrict__ Wl, const float* __restrict__ Wr,
                                    int dout, int ntiles, unsigned short* __restrict__ packed) {
  int frag  = blockIdx.x;                 // ktile * ntiles + ntile
  int ktile = frag / ntiles;
  int ntile = frag - ktile * ntiles;
  int l     = threadIdx.x;                // 0..31
  int col   = ntile * 16 + (l & 15);
  int kbase = ktile * 32 + (l >> 4) * 16;
  unsigned short* out = packed + (size_t)frag * 512 + l * 16;
  #pragma unroll
  for (int i = 0; i < 16; ++i) {
    int k = kbase + i;
    float v = 0.0f;
    if (col < dout) v = (k < D_FEAT) ? Wl[k * dout + col] : Wr[(k - D_FEAT) * dout + col];
    out[i] = f2bf(v);
  }
}

// ---------------- fused SAGE layer GEMM: out = [mean|x] @ [Wl;Wr] + b ----------------
// one wave per 16-node tile; bf16 WMMA, f32 accumulate; B frags prefetched per ktile
template <int NTILES, int DOUT, bool RELU>
__global__ __launch_bounds__(32) void sage_gemm_kernel(
    const float* __restrict__ xin, const float* __restrict__ mean,
    const unsigned short* __restrict__ packedB,
    const float* __restrict__ bias, float* __restrict__ out) {
  __shared__ unsigned short Alds[16 * A_STRIDE];

  const int tile = blockIdx.x;
  const int lane = threadIdx.x;
  const int nodeBase = tile * 16;

  // stage A = [mean | x] as bf16 rows (K=256) into LDS
  for (int idx = lane; idx < 16 * 32; idx += 32) {   // (row, float4-chunk)
    int row = idx >> 5;
    int c4  = idx & 31;
    int node = nodeBase + row;
    float4 ma = ((const float4*)(mean + (size_t)node * D_FEAT))[c4];
    float4 xa = ((const float4*)(xin  + (size_t)node * D_FEAT))[c4];
    unsigned int m0 = (unsigned int)f2bf(ma.x) | ((unsigned int)f2bf(ma.y) << 16);
    unsigned int m1 = (unsigned int)f2bf(ma.z) | ((unsigned int)f2bf(ma.w) << 16);
    unsigned int x0 = (unsigned int)f2bf(xa.x) | ((unsigned int)f2bf(xa.y) << 16);
    unsigned int x1 = (unsigned int)f2bf(xa.z) | ((unsigned int)f2bf(xa.w) << 16);
    unsigned int* lm = (unsigned int*)&Alds[row * A_STRIDE + 4 * c4];
    lm[0] = m0; lm[1] = m1;
    unsigned int* lx = (unsigned int*)&Alds[row * A_STRIDE + D_FEAT + 4 * c4];
    lx[0] = x0; lx[1] = x1;
  }
  __syncthreads();

  const int rowA  = lane & 15;
  const int khalf = lane >> 4;

  v8f acc[NTILES];
  #pragma unroll
  for (int nt = 0; nt < NTILES; ++nt) {
    int col = nt * 16 + (lane & 15);
    float b = (col < DOUT) ? bias[col] : 0.0f;
    #pragma unroll
    for (int j = 0; j < 8; ++j) acc[nt][j] = b;
  }

  const unsigned int* lds32 = (const unsigned int*)Alds;
  #pragma unroll
  for (int kt = 0; kt < K_TILES; ++kt) {
    // A fragment: lane (row, khalf) gathers K pairs {0..15}+khalf*16 of this ktile
    Frag16 a;
    #pragma unroll
    for (int i = 0; i < 8; ++i) {
      int kp = kt * 32 + khalf * 16 + 2 * i;
      a.u[i] = lds32[rowA * (A_STRIDE / 2) + (kp >> 1)];
    }
    // prefetch ALL B fragments of this ktile, then issue WMMAs back-to-back
    Frag16 bfs[NTILES];
    #pragma unroll
    for (int nt = 0; nt < NTILES; ++nt) {
      const uint4* bp = (const uint4*)(packedB + ((size_t)(kt * NTILES + nt) * 512 + lane * 16));
      bfs[nt].q[0] = bp[0];
      bfs[nt].q[1] = bp[1];
    }
    #pragma unroll
    for (int nt = 0; nt < NTILES; ++nt) {
      acc[nt] = __builtin_amdgcn_wmma_f32_16x16x32_bf16(
          false, a.v, false, bfs[nt].v, (short)0, acc[nt], false, false);
    }
  }

  // epilogue: C/D layout -> element (M = khalf*8 + j, N = lane&15)
  #pragma unroll
  for (int nt = 0; nt < NTILES; ++nt) {
    int col = nt * 16 + (lane & 15);
    if (col < DOUT) {
      #pragma unroll
      for (int j = 0; j < 8; ++j) {
        float v = acc[nt][j];
        if (RELU) v = v > 0.0f ? v : 0.0f;
        out[(size_t)(nodeBase + khalf * 8 + j) * DOUT + col] = v;
      }
    }
  }
}

// ---------------- in-place row-wise log_softmax over 47 classes ----------------
__global__ void log_softmax_kernel(float* __restrict__ y) {
  int node = blockIdx.x * (blockDim.x >> 5) + (threadIdx.x >> 5);
  int lane = threadIdx.x & 31;
  if (node >= N_NODES) return;
  float* row = y + (size_t)node * D_CLS;
  float v0 = (lane      < D_CLS) ? row[lane]      : -__builtin_inff();
  float v1 = (lane + 32 < D_CLS) ? row[lane + 32] : -__builtin_inff();
  float m = fmaxf(v0, v1);
  #pragma unroll
  for (int off = 16; off > 0; off >>= 1) m = fmaxf(m, __shfl_xor(m, off, 32));
  float e0 = (lane      < D_CLS) ? __expf(v0 - m) : 0.0f;
  float e1 = (lane + 32 < D_CLS) ? __expf(v1 - m) : 0.0f;
  float s = e0 + e1;
  #pragma unroll
  for (int off = 16; off > 0; off >>= 1) s += __shfl_xor(s, off, 32);
  float lse = m + __logf(s);
  if (lane      < D_CLS) row[lane]      = v0 - lse;
  if (lane + 32 < D_CLS) row[lane + 32] = v1 - lse;
}

extern "C" void kernel_launch(void* const* d_in, const int* in_sizes, int n_in,
                              void* d_out, int out_size, void* d_ws, size_t ws_size,
                              hipStream_t stream) {
  const float*     x   = (const float*)d_in[0];
  const long long* ei  = (const long long*)d_in[1];   // int64 (2, E)
  const long long* src = ei;
  const long long* dst = ei + N_EDGES;
  const float* Wl0 = (const float*)d_in[2];
  const float* Wr0 = (const float*)d_in[3];
  const float* b0  = (const float*)d_in[4];
  const float* Wl1 = (const float*)d_in[5];
  const float* Wr1 = (const float*)d_in[6];
  const float* b1  = (const float*)d_in[7];
  const float* Wl2 = (const float*)d_in[8];
  const float* Wr2 = (const float*)d_in[9];
  const float* b2  = (const float*)d_in[10];

  // workspace layout
  char* ws = (char*)d_ws;
  size_t off = 0;
  auto take = [&](size_t bytes) -> void* {
    void* p = ws + off;
    off = (off + bytes + 255) & ~(size_t)255;
    return p;
  };
  int*   deg    = (int*)take((size_t)N_NODES * 4);
  int*   offs   = (int*)take((size_t)(N_NODES + 1) * 4);
  int*   cursor = (int*)take((size_t)N_NODES * 4);
  int*   csr    = (int*)take((size_t)N_EDGES * 4);
  float* mean   = (float*)take((size_t)N_NODES * D_FEAT * 4);
  float* h1     = (float*)take((size_t)N_NODES * D_FEAT * 4);
  float* h2     = (float*)take((size_t)N_NODES * D_FEAT * 4);
  unsigned short* pb0 = (unsigned short*)take((size_t)K_TILES * 8 * 512 * 2);
  unsigned short* pb1 = (unsigned short*)take((size_t)K_TILES * 8 * 512 * 2);
  unsigned short* pb2 = (unsigned short*)take((size_t)K_TILES * 3 * 512 * 2);
  float* outf = (float*)d_out;

  const int nTiles    = N_NODES / 16;                    // 3125 exactly
  const int aggBlocks = (N_NODES + 7) / 8;               // 8 waves per 256-thread block

  // ---- CSR build (reused by all 3 layers) ----
  hipMemsetAsync(deg, 0, (size_t)N_NODES * 4, stream);
  count_edges_kernel<<<(N_EDGES + 255) / 256, 256, 0, stream>>>(dst, deg);
  scan_offsets_kernel<<<1, SCAN_T, 0, stream>>>(deg, offs);
  copy_int_kernel<<<(N_NODES + 255) / 256, 256, 0, stream>>>(offs, cursor, N_NODES);
  fill_csr_kernel<<<(N_EDGES + 255) / 256, 256, 0, stream>>>(src, dst, cursor, csr);

  // ---- pre-swizzle all layer weights into WMMA B-fragment layout ----
  pack_weights_kernel<<<K_TILES * 8, 32, 0, stream>>>(Wl0, Wr0, D_FEAT, 8, pb0);
  pack_weights_kernel<<<K_TILES * 8, 32, 0, stream>>>(Wl1, Wr1, D_FEAT, 8, pb1);
  pack_weights_kernel<<<K_TILES * 3, 32, 0, stream>>>(Wl2, Wr2, D_CLS, 3, pb2);

  // ---- layer 0 ----
  aggregate_kernel<<<aggBlocks, 256, 0, stream>>>(x, csr, offs, mean);
  sage_gemm_kernel<8, D_FEAT, true><<<nTiles, 32, 0, stream>>>(x, mean, pb0, b0, h1);

  // ---- layer 1 ----
  aggregate_kernel<<<aggBlocks, 256, 0, stream>>>(h1, csr, offs, mean);
  sage_gemm_kernel<8, D_FEAT, true><<<nTiles, 32, 0, stream>>>(h1, mean, pb1, b1, h2);

  // ---- layer 2 (no relu) -> d_out, then in-place log_softmax ----
  aggregate_kernel<<<aggBlocks, 256, 0, stream>>>(h2, csr, offs, mean);
  sage_gemm_kernel<3, D_CLS, false><<<nTiles, 32, 0, stream>>>(h2, mean, pb2, b2, outf);

  log_softmax_kernel<<<(N_NODES + 7) / 8, 256, 0, stream>>>(outf);
}
